// MultiHeadAttention4SparseKT_21320217657942
// MI455X (gfx1250) — matrix operand order repaired and checked by reading, not compiled, not run on hardware
//
#include <hip/hip_runtime.h>
#include <hip/hip_fp16.h>

// MultiHeadAttention with top-5 sparse re-softmax, MI455X (gfx1250, wave32, WMMA).
// B=16 S=512 DM=256 H=8 DF=32, K_INDEX=5.

typedef __attribute__((ext_vector_type(16))) _Float16 v16h;
typedef __attribute__((ext_vector_type(8)))  float    v8f;
typedef __attribute__((ext_vector_type(4)))  float    f32x4;
typedef __attribute__((ext_vector_type(4)))  int      i32x4;

#define B_   16
#define S_   512
#define DM_  256
#define H_   8
#define DF_  32
#define NEGM (-1.0e32f)

#define AS1 __attribute__((address_space(1)))
#define AS3 __attribute__((address_space(3)))

#if defined(__has_builtin)
#if __has_builtin(__builtin_amdgcn_global_load_async_to_lds_b128)
#define USE_ASYNC_LDS 1
#endif
#endif

union V16 { v16h v; uint4 q[2]; };
union H2  { _Float16 h[2]; unsigned int u; };

__device__ __forceinline__ float wave_max(float v) {
  #pragma unroll
  for (int m = 16; m >= 1; m >>= 1) v = fmaxf(v, __shfl_xor(v, m, 32));
  return v;
}
__device__ __forceinline__ float wave_sum(float v) {
  #pragma unroll
  for (int m = 16; m >= 1; m >>= 1) v += __shfl_xor(v, m, 32);
  return v;
}

// ---------------------------------------------------------------------------
// Pack W (row-major [n][k], y = x @ W^T) into WMMA f16 B-operand layout:
// tile (nt,kt): lane L (g=L/16, n=nt*16+L%16), dword v: k = kt*32 + (v<4 ? g*8 + 2v : 16 + g*8 + 2(v-4)).
// packed dword index = ((nt*8+kt)*32 + lane)*8 + v  (== flat thread id).
// z: 0=Wk, 1=Wv, 2=Wo.
// ---------------------------------------------------------------------------
__global__ __launch_bounds__(256) void k_pack(const float* __restrict__ Wk,
                                              const float* __restrict__ Wv,
                                              const float* __restrict__ Wo,
                                              unsigned int* __restrict__ P) {
  int id = blockIdx.x * 256 + threadIdx.x;            // 0..32767
  const float* W = (blockIdx.z == 0) ? Wk : ((blockIdx.z == 1) ? Wv : Wo);
  unsigned int* out = P + (size_t)blockIdx.z * 32768;
  int v = id & 7, lane = (id >> 3) & 31, kt = (id >> 8) & 7, nt = id >> 11;
  int n = nt * 16 + (lane & 15), g = lane >> 4, k = kt * 32 + (v >> 2) * 16 + g * 8 + 2 * (v & 3);
  H2 h;
  h.h[0] = (_Float16)W[(size_t)n * DM_ + k];
  h.h[1] = (_Float16)W[(size_t)n * DM_ + k + 1];
  out[id] = h.u;
}

// ---------------------------------------------------------------------------
// Input projections: Y = X @ W^T + b, stored f16 head-major [B,H,S,DF].
// One wave per 16x16 output tile; 8 WMMA steps over K=256. Streaming f32
// inputs are read with non-temporal b128 loads (read-once data).
// z: 0 -> Qh = q@Wk^T+bk, 1 -> Kh = k@Wk^T+bk, 2 -> Vh = v@Wv^T+bv.
// ---------------------------------------------------------------------------
__global__ __launch_bounds__(32) void k_proj(const float* __restrict__ q,
                                             const float* __restrict__ k,
                                             const float* __restrict__ v,
                                             const unsigned int* __restrict__ packs,
                                             const float* __restrict__ bk,
                                             const float* __restrict__ bvv,
                                             _Float16* __restrict__ Qh,
                                             _Float16* __restrict__ Kh,
                                             _Float16* __restrict__ Vh) {
  int lane = threadIdx.x, g = lane >> 4, l15 = lane & 15;
  int sel = blockIdx.z;
  const float* x = (sel == 0) ? q : ((sel == 1) ? k : v);
  const unsigned int* pack = packs + (size_t)((sel == 2) ? 1 : 0) * 32768;
  const float* bias = (sel == 2) ? bvv : bk;
  _Float16* out = (sel == 0) ? Qh : ((sel == 1) ? Kh : Vh);
  int mbase = blockIdx.x * 16, nt = blockIdx.y, nbase = nt * 16;

  v8f c = {};
  const float* xrow = x + (size_t)(mbase + l15) * DM_;
  for (int kt = 0; kt < 8; ++kt) {
    int kb = kt * 32;
    const f32x4* x0 = (const f32x4*)(xrow + kb + g * 8);
    const f32x4* x1 = (const f32x4*)(xrow + kb + 16 + g * 8);
    f32x4 a0 = __builtin_nontemporal_load(x0);
    f32x4 a1 = __builtin_nontemporal_load(x0 + 1);
    f32x4 a2 = __builtin_nontemporal_load(x1);
    f32x4 a3 = __builtin_nontemporal_load(x1 + 1);
    V16 a;
    #pragma unroll
    for (int e = 0; e < 4; ++e) {
      a.v[e]      = (_Float16)a0[e];
      a.v[4 + e]  = (_Float16)a1[e];
      a.v[8 + e]  = (_Float16)a2[e];
      a.v[12 + e] = (_Float16)a3[e];
    }
    V16 bf;
    const uint4* pw = ((const uint4*)pack) + ((size_t)(nt * 8 + kt) * 32 + lane) * 2;
    bf.q[0] = pw[0];
    bf.q[1] = pw[1];
    c = __builtin_amdgcn_wmma_f32_16x16x32_f16(false, a.v, false, bf.v,
                                               (short)0, c, false, false);
  }
  int n = nbase + l15, h = n >> 5, d = n & 31;
  float bb = bias[n];
  #pragma unroll
  for (int r = 0; r < 8; ++r) {
    int mg = mbase + r + 8 * g;
    int bidx = mg >> 9, s = mg & (S_ - 1);
    out[((size_t)(bidx * H_ + h) * S_ + s) * DF_ + d] = (_Float16)(c[r] + bb);
  }
}

// ---------------------------------------------------------------------------
// Attention: one block per (b,h, q-chunk). 4 waves, each owns one 16-row
// q-tile. K for this (b,h) staged into LDS with async global->LDS b128
// copies (ASYNCcnt). Scores via WMMA into per-wave LDS scratch; per-row
// softmax + top-5 threshold + re-softmax; attn rows written with
// non-temporal b128 stores (write-once 134MB stream); sparse ctx from the
// <=16 surviving (index,weight) pairs.
// ---------------------------------------------------------------------------
__global__ __launch_bounds__(128) void k_attn(const _Float16* __restrict__ Qh,
                                              const _Float16* __restrict__ Kh,
                                              const _Float16* __restrict__ Vh,
                                              const int* __restrict__ zero_pad,
                                              float* __restrict__ attn_out,
                                              _Float16* __restrict__ Ctx) {
  extern __shared__ char smem[];
  _Float16* Ksh   = (_Float16*)smem;                         // 32768 B
  float*    scrA  = (float*)(smem + 32768);                  // 4 * 16*512*4 B
  int*      idxA  = (int*)(smem + 32768 + 131072);           // 4 * 16
  float*    wA    = (float*)(smem + 32768 + 131072 + 256);   // 4 * 16
  int*      cntA  = (int*)(smem + 32768 + 131072 + 512);     // 4

  int tid = threadIdx.x, lane = tid & 31, w = tid >> 5;
  int bh = blockIdx.x, b = bh >> 3, h = bh & 7;

  // ---- K staging: global -> LDS, 512x32 f16 = 2048 x b128 ----
#ifdef USE_ASYNC_LDS
  {
    AS1 i32x4* gk = (AS1 i32x4*)(unsigned long long)
                    (const char*)(Kh + (size_t)bh * S_ * DF_);
    // generic LDS pointer: low 32 bits == LDS byte offset (aperture rule)
    AS3 i32x4* lk = (AS3 i32x4*)(unsigned int)(unsigned long long)(char*)Ksh;
    for (int i = tid; i < 2048; i += 128)
      __builtin_amdgcn_global_load_async_to_lds_b128(gk + i, lk + i, 0, 0);
  }
#if defined(__has_builtin) && __has_builtin(__builtin_amdgcn_s_wait_asynccnt)
  __builtin_amdgcn_s_wait_asynccnt(0);
#else
  asm volatile("s_wait_asynccnt 0" ::: "memory");
#endif
#else
  {
    const uint4* ksrc = (const uint4*)(Kh + (size_t)bh * S_ * DF_);
    uint4* kdst = (uint4*)Ksh;
    for (int i = tid; i < 2048; i += 128) kdst[i] = ksrc[i];
  }
#endif
  __syncthreads();

  float* scr = scrA + w * (16 * 512);
  int*   idxb = idxA + w * 16;
  float* wb   = wA + w * 16;
  int*   cnt  = cntA + w;

  int qt = blockIdx.y * 4 + w, qbase = qt * 16;
  int g = lane >> 4, l15 = lane & 15;

  // Q fragment (A-matrix, f16 16x32): two b128 loads per lane
  V16 a;
  const _Float16* qrow = Qh + ((size_t)bh * S_ + qbase + l15) * DF_;
  a.q[0] = *(const uint4*)(qrow + g * 8);
  a.q[1] = *(const uint4*)(qrow + 16 + g * 8);

  const float scale = 0.17677669529663687f;  // 1/sqrt(DF)

  // Phase 1: scores for all 32 k-tiles (skip fully-masked upper triangle)
  for (int j = 0; j < 32; ++j) {
    int kb = j * 16;
    if (kb > qbase + 15) {
      #pragma unroll
      for (int r = 0; r < 8; ++r) scr[(r + 8 * g) * 512 + kb + l15] = NEGM;
      continue;
    }
    V16 bf;  // B = K^T tile (32x16): lane col = l15, k = g*8 + 0..7, 16 + g*8 + 0..7
    const _Float16* krow = Ksh + (size_t)(kb + l15) * DF_ + g * 8;
    bf.q[0] = *(const uint4*)(krow);
    bf.q[1] = *(const uint4*)(krow + 16);
    v8f c = {};
    c = __builtin_amdgcn_wmma_f32_16x16x32_f16(false, a.v, false, bf.v,
                                               (short)0, c, false, false);
    #pragma unroll
    for (int r = 0; r < 8; ++r) {
      int row = qbase + r + 8 * g, col = kb + l15;
      float s = c[r] * scale;
      if (col > row) s = NEGM;
      scr[(r + 8 * g) * 512 + col] = s;
    }
  }
  asm volatile("s_wait_dscnt 0" ::: "memory");  // wave-internal LDS WAR/RAW fence
  __builtin_amdgcn_wave_barrier();

  int zp = zero_pad[0];
  float* aout = attn_out + (size_t)bh * S_ * S_;

  // Phase 2: per-row softmax / top-5 / output / sparse ctx
  for (int r16 = 0; r16 < 16; ++r16) {
    int qi = qbase + r16;
    float vals[16];
    #pragma unroll
    for (int t = 0; t < 16; ++t) vals[t] = scr[r16 * 512 + lane * 16 + t];

    float lm = -3.0e38f;
    #pragma unroll
    for (int t = 0; t < 16; ++t) lm = fmaxf(lm, vals[t]);
    float rmax = wave_max(lm);

    float ls = 0.f;
    #pragma unroll
    for (int t = 0; t < 16; ++t) ls += expf(vals[t] - rmax);
    float Z = wave_sum(ls);
    float rZ = 1.0f / Z;  // == attn1 at the row max (exp(0)/Z)

    bool sparse = (qi >= 6);  // K_INDEX+1
    float thr = -3.0e38f, norm = 1.0f;
    if (sparse) {
      float prev = 3.0e38f;  // iterative distinct-max: 5th largest score
      for (int it = 0; it < 5; ++it) {
        float lmx = -3.0e38f;
        #pragma unroll
        for (int t = 0; t < 16; ++t) {
          float vv = vals[t];
          if (vv < prev) lmx = fmaxf(lmx, vv);
        }
        prev = wave_max(lmx);
      }
      thr = prev;
      float lse = 0.f;  // sum over survivors of exp(attn1_i - attn1_max)
      #pragma unroll
      for (int t = 0; t < 16; ++t) {
        float vv = vals[t];
        if (vv >= thr) lse += expf(expf(vv - rmax) * rZ - rZ);
      }
      norm = 1.0f / wave_sum(lse);
    }

    bool zero = (qi == 0) && (zp != 0);
    if (lane == 0) *cnt = 0;
    float wrow[16];
    #pragma unroll
    for (int t = 0; t < 16; ++t) {
      int col = lane * 16 + t;
      float vv = vals[t], wout = 0.f;
      if (!zero) {
        if (sparse) {
          if (vv >= thr) wout = expf(expf(vv - rmax) * rZ - rZ) * norm;
        } else {
          wout = expf(vv - rmax) * rZ;  // dense softmax; masked -> exactly 0
        }
      }
      wrow[t] = wout;
      if (wout > 0.f) {
        int pos = atomicAdd(cnt, 1);
        if (pos < 16) { idxb[pos] = col; wb[pos] = wout; }
      }
    }
    // attn row: write-once stream -> non-temporal b128 stores
    {
      f32x4* orow4 = (f32x4*)(aout + (size_t)qi * 512 + lane * 16);
      #pragma unroll
      for (int t4 = 0; t4 < 4; ++t4) {
        f32x4 pk = {wrow[4 * t4], wrow[4 * t4 + 1],
                    wrow[4 * t4 + 2], wrow[4 * t4 + 3]};
        __builtin_nontemporal_store(pk, orow4 + t4);
      }
    }
    __builtin_amdgcn_wave_barrier();
    asm volatile("s_wait_dscnt 0" ::: "memory");

    int nsv = *cnt; nsv = nsv > 16 ? 16 : nsv;
    float acc = 0.f;  // lane = head-dim d
    const _Float16* vbase = Vh + (size_t)bh * S_ * DF_;
    for (int jj = 0; jj < nsv; ++jj)
      acc += wb[jj] * (float)vbase[(size_t)idxb[jj] * DF_ + lane];
    Ctx[((size_t)(b * S_ + qi)) * DM_ + h * DF_ + lane] = (_Float16)acc;
    __builtin_amdgcn_wave_barrier();
  }
}

// ---------------------------------------------------------------------------
// Output projection: out = Ctx(f16) @ Wo^T + bo, f32 result (non-temporal).
// ---------------------------------------------------------------------------
__global__ __launch_bounds__(32) void k_out(const _Float16* __restrict__ Ctx,
                                            const unsigned int* __restrict__ WoPack,
                                            const float* __restrict__ bo,
                                            float* __restrict__ out) {
  int lane = threadIdx.x, g = lane >> 4, l15 = lane & 15;
  int mbase = blockIdx.x * 16, nt = blockIdx.y, nbase = nt * 16;
  v8f c = {};
  const _Float16* xrow = Ctx + (size_t)(mbase + l15) * DM_;
  for (int kt = 0; kt < 8; ++kt) {
    int kb = kt * 32;
    V16 a;
    a.q[0] = *(const uint4*)(xrow + kb + g * 8);
    a.q[1] = *(const uint4*)(xrow + kb + 16 + g * 8);
    V16 bf;
    const uint4* pw = ((const uint4*)WoPack) + ((size_t)(nt * 8 + kt) * 32 + lane) * 2;
    bf.q[0] = pw[0];
    bf.q[1] = pw[1];
    c = __builtin_amdgcn_wmma_f32_16x16x32_f16(false, a.v, false, bf.v,
                                               (short)0, c, false, false);
  }
  float bb = bo[nbase + l15];
  #pragma unroll
  for (int r = 0; r < 8; ++r) {
    int mg = mbase + r + 8 * g;
    __builtin_nontemporal_store(c[r] + bb, &out[(size_t)mg * DM_ + nbase + l15]);
  }
}

// ---------------------------------------------------------------------------
extern "C" void kernel_launch(void* const* d_in, const int* in_sizes, int n_in,
                              void* d_out, int out_size, void* d_ws, size_t ws_size,
                              hipStream_t stream) {
  (void)in_sizes; (void)n_in; (void)out_size; (void)ws_size;
  const float* q  = (const float*)d_in[0];
  const float* k  = (const float*)d_in[1];
  const float* v  = (const float*)d_in[2];
  // d_in[3] = mask (tril; causal mask is applied analytically)
  const int*   zp = (const int*)d_in[4];
  const float* Wk = (const float*)d_in[5];
  const float* bk = (const float*)d_in[6];
  const float* Wv = (const float*)d_in[7];
  const float* bv = (const float*)d_in[8];
  const float* Wo = (const float*)d_in[9];
  const float* bo = (const float*)d_in[10];

  float* out  = (float*)d_out;                       // [B,S,DM]
  float* attn = out + (size_t)B_ * S_ * DM_;         // [B,H,S,S]

  char* ws = (char*)d_ws;
  unsigned int* packs = (unsigned int*)ws;                       //  384 KB (3 packed W)
  _Float16* Qh  = (_Float16*)(ws + 393216);                      //  4 MB
  _Float16* Kh  = (_Float16*)(ws + 393216 + 1 * 4194304);        //  4 MB
  _Float16* Vh  = (_Float16*)(ws + 393216 + 2 * 4194304);        //  4 MB
  _Float16* Ctx = (_Float16*)(ws + 393216 + 3 * 4194304);        //  4 MB

  k_pack<<<dim3(128, 1, 3), 256, 0, stream>>>(Wk, Wv, Wo, packs);
  k_proj<<<dim3(512, 16, 3), 32, 0, stream>>>(q, k, v, packs, bk, bv, Qh, Kh, Vh);
  size_t smem = 32768 + 131072 + 256 + 256 + 16;  // K + 4x score scratch + pair bufs
  k_attn<<<dim3(128, 8), 128, smem, stream>>>(Qh, Kh, Vh, zp, attn, Ctx);
  k_out<<<dim3(512, 16), 32, 0, stream>>>(Ctx, packs + 2 * 32768, bo, out);
}
